// VisualBranch_vsgnet_72438918414396
// MI455X (gfx1250) — matrix operand order repaired and checked by reading, not compiled, not run on hardware
//
#include <hip/hip_runtime.h>
#include <stdint.h>

// ---------------------------------------------------------------------------
// VisualBranch fused pipeline for MI455X (gfx1250, wave32, WMMA).
//   M1  = W1 @ W2                      (bf16 WMMA, stored transposed [512,2048])
//   Q   = OBJ @ M1_top                 (bf16 WMMA, f32 out [2048,512])
//   Cc  = CTX @ M1_bot + b1@W2 + b2    (tiny, VALU)
//   out = relu(0.5*(Q[g0]+Q[g1]) + Cc[b])
// All GEMM B-operands are kept K-contiguous ([N,K]) in global memory so
// global->LDS staging is pure 16B copies -> GLOBAL_LOAD_ASYNC_TO_LDS_B128
// double-buffered pipeline (ASYNCcnt), feeding v_wmma_f32_16x16x32_bf16.
// ---------------------------------------------------------------------------

typedef __bf16 bf16;
typedef __attribute__((ext_vector_type(16))) __bf16 v16bf;
typedef __attribute__((ext_vector_type(8)))  __bf16 v8bf;
typedef __attribute__((ext_vector_type(8)))  float  v8f;
typedef __attribute__((ext_vector_type(4)))  int    i32x4;

#define OBJS 32
#define BDIM 64
#define RREL 256
#define DDIM 1024
#define NOUT 512

#if defined(__gfx1250__) && __has_builtin(__builtin_amdgcn_global_load_async_to_lds_b128)
#define HAVE_ASYNC 1
#else
#define HAVE_ASYNC 0
#endif

static __device__ __forceinline__ bf16 f2bf(float f) {
  unsigned int u = __builtin_bit_cast(unsigned int, f);
  unsigned int r = (u + 0x7FFFu + ((u >> 16) & 1u)) >> 16;  // RNE
  return __builtin_bit_cast(bf16, (unsigned short)r);
}
static __device__ __forceinline__ float bf2f(bf16 h) {
  unsigned int u = ((unsigned int)__builtin_bit_cast(unsigned short, h)) << 16;
  return __builtin_bit_cast(float, u);
}

// 16-byte global -> LDS copy: async (ASYNCcnt) when available.
static __device__ __forceinline__ void cp16(void* lds, const void* g) {
#if HAVE_ASYNC
  i32x4* gg = (i32x4*)(void*)g;   // strip const, generic typed ptr
  i32x4* ll = (i32x4*)lds;
  __builtin_amdgcn_global_load_async_to_lds_b128(
      (__attribute__((address_space(1))) i32x4*)gg,   // global (AS1)
      (__attribute__((address_space(3))) i32x4*)ll,   // LDS (AS3)
      0, 0);
#else
  *(v8bf*)lds = *(const v8bf*)g;
#endif
}
static __device__ __forceinline__ void wait_async0() {
#if HAVE_ASYNC
#if __has_builtin(__builtin_amdgcn_s_wait_asynccnt)
  __builtin_amdgcn_s_wait_asynccnt(0);
#else
  asm volatile("s_wait_asynccnt 0x0" ::: "memory");
#endif
#endif
}

// ---------------------------------------------------------------------------
// f32 -> bf16 bulk convert (n multiple of 1024 per launch config)
// ---------------------------------------------------------------------------
__global__ void cvt_f32_bf16(const float* __restrict__ src,
                             bf16* __restrict__ dst, int n) {
  int i = (blockIdx.x * blockDim.x + threadIdx.x) * 4;
  if (i + 3 < n) {
    float4 v = *(const float4*)(src + i);
    dst[i + 0] = f2bf(v.x);
    dst[i + 1] = f2bf(v.y);
    dst[i + 2] = f2bf(v.z);
    dst[i + 3] = f2bf(v.w);
  }
}

// src [rows, cols] f32 -> dst [cols, rows] bf16 (one-time weight re-pack)
__global__ void cvt_transpose_f32_bf16(const float* __restrict__ src,
                                       bf16* __restrict__ dst, int rows,
                                       int cols) {
  int i = blockIdx.x * blockDim.x + threadIdx.x;
  if (i < rows * cols) {
    int r = i / cols, c = i - r * cols;
    dst[(size_t)c * rows + r] = f2bf(src[i]);
  }
}

// ---------------------------------------------------------------------------
// C = A[MxK] * Bt[NxK]^T, bf16 inputs, f32 WMMA accumulate.
// Block tile BM=128 x BN=64 x BK=64, 256 threads = 8 waves (4xM, 2xN),
// wave owns 32x32 = 2x2 WMMA 16x16 tiles. Both LDS tiles are K-contiguous
// (stride LDK=72 bf16 to dodge bank conflicts), staged by double-buffered
// 16B async copies; fragments read with paired ds_load_b128 per the
// ISA 7.12.2 wave32 layouts.
// OUT_TRANS_BF16: store C transposed as bf16 [N, ldout] (contiguous v8bf per
// tile-lane); else plain f32 [M, N].
// ---------------------------------------------------------------------------
#define BM 128
#define BN 64
#define BK 64
#define LDK 72

template <bool OUT_TRANS_BF16>
__global__ __launch_bounds__(256) void gemm_bf16_wmma(
    const bf16* __restrict__ A, int lda, const bf16* __restrict__ Bt, int ldb,
    void* __restrict__ Cout, int ldout, int M, int N, int K) {
  __shared__ __align__(16) bf16 Asm[2][BM * LDK];
  __shared__ __align__(16) bf16 Bsm[2][BN * LDK];

  const int tid  = threadIdx.x;
  const int lane = tid & 31;
  const int wid  = tid >> 5;
  const int wm   = (wid & 3) * 32;   // wave row offset in block tile
  const int wn   = (wid >> 2) * 32;  // wave col offset in block tile
  const int m0   = blockIdx.y * BM;
  const int n0   = blockIdx.x * BN;
  const int lrow  = lane & 15;
  const int lhalf = lane >> 4;

  v8f acc[2][2] = {};

  // stage tile at K-offset kk into buffer buf (4+2 = 6 x 16B per thread)
  auto stage = [&](int buf, int kk) {
#pragma unroll
    for (int r = 0; r < 4; ++r) {        // A: 128 rows x 8 chunks
      int c = tid + r * 256;
      int m = c >> 3, h = (c & 7) * 8;
      cp16(&Asm[buf][m * LDK + h], A + (size_t)(m0 + m) * lda + kk + h);
    }
#pragma unroll
    for (int r = 0; r < 2; ++r) {        // B: 64 rows x 8 chunks
      int c = tid + r * 256;
      int n = c >> 3, h = (c & 7) * 8;
      cp16(&Bsm[buf][n * LDK + h], Bt + (size_t)(n0 + n) * ldb + kk + h);
    }
  };

  stage(0, 0);
  for (int kk = 0; kk < K; kk += BK) {
    const int cur = (kk >> 6) & 1;
    wait_async0();           // this wave's copies into cur complete
    __syncthreads();         // whole block's copies complete / prev reads done
    if (kk + BK < K) stage(cur ^ 1, kk + BK);  // overlap next copies w/ math

#pragma unroll
    for (int s = 0; s < 2; ++s) {        // two K=32 steps per BK=64 tile
      v16bf afrag[2], bfrag[2];
#pragma unroll
      for (int t = 0; t < 2; ++t) {
        const bf16* ap = &Asm[cur][(wm + t * 16 + lrow) * LDK + s * 32 + lhalf * 8];
        v8bf alo = *(const v8bf*)ap;
        v8bf ahi = *(const v8bf*)(ap + 16);
        afrag[t] = __builtin_shufflevector(alo, ahi, 0, 1, 2, 3, 4, 5, 6, 7,
                                           8, 9, 10, 11, 12, 13, 14, 15);
        const bf16* bp = &Bsm[cur][(wn + t * 16 + lrow) * LDK + s * 32 + lhalf * 8];
        v8bf blo = *(const v8bf*)bp;
        v8bf bhi = *(const v8bf*)(bp + 16);
        bfrag[t] = __builtin_shufflevector(blo, bhi, 0, 1, 2, 3, 4, 5, 6, 7,
                                           8, 9, 10, 11, 12, 13, 14, 15);
      }
#pragma unroll
      for (int tm = 0; tm < 2; ++tm)
#pragma unroll
        for (int tn = 0; tn < 2; ++tn)
          acc[tm][tn] = __builtin_amdgcn_wmma_f32_16x16x32_bf16(
              false, afrag[tm], false, bfrag[tn], (short)0, acc[tm][tn],
              false, false);
    }
    __syncthreads();
  }

  // Epilogue. C/D layout: VGPR v -> row = v + 8*(lane/16), col = lane%16.
#pragma unroll
  for (int tm = 0; tm < 2; ++tm)
#pragma unroll
    for (int tn = 0; tn < 2; ++tn) {
      int row0 = m0 + wm + tm * 16 + lhalf * 8;
      int col  = n0 + wn + tn * 16 + lrow;
      if (OUT_TRANS_BF16) {
        v8bf ov;
#pragma unroll
        for (int v = 0; v < 8; ++v) ov[v] = f2bf(acc[tm][tn][v]);
        *(v8bf*)((bf16*)Cout + (size_t)col * ldout + row0) = ov;  // 16B store
      } else {
#pragma unroll
        for (int v = 0; v < 8; ++v)
          ((float*)Cout)[(size_t)(row0 + v) * ldout + col] = acc[tm][tn][v];
      }
    }
  (void)M; (void)N;
}

// ---------------------------------------------------------------------------
// Cc[b,n] = b2[n] + sum_k b1[k]*W2[k,n] + sum_k ctx[b,k]*M1t[n, 1024+k]
// ---------------------------------------------------------------------------
__global__ void ctx_bias_kernel(const float* __restrict__ ctx,
                                const float* __restrict__ b1,
                                const float* __restrict__ W2,
                                const float* __restrict__ b2,
                                const bf16* __restrict__ M1t,
                                float* __restrict__ Cc) {
  int idx = blockIdx.x * blockDim.x + threadIdx.x;  // [0, 64*512)
  int b = idx >> 9;
  int n = idx & (NOUT - 1);
  float acc = b2[n];
  for (int k = 0; k < DDIM; ++k) acc += b1[k] * W2[k * NOUT + n];
  const float* cb = ctx + b * DDIM;
  const bf16* mb  = M1t + (size_t)n * (2 * DDIM) + DDIM;  // M1_bot, contiguous
  for (int k = 0; k < DDIM; ++k) acc += cb[k] * bf2f(mb[k]);
  Cc[idx] = acc;
}

// ---------------------------------------------------------------------------
// out[row,:] = relu(0.5*(Q[g0]+Q[g1]) + Cc[b]) ; one block per output row.
// ---------------------------------------------------------------------------
__global__ void gather_relu_out(const int* __restrict__ pairs,
                                const float* __restrict__ Q,
                                const float* __restrict__ Cc,
                                float* __restrict__ out) {
  int row = blockIdx.x;            // 0..16383
  int b   = row >> 8;
  int g0  = pairs[row * 2 + 0] + b * OBJS;
  int g1  = pairs[row * 2 + 1] + b * OBJS;
  int c   = threadIdx.x * 4;       // 128 threads * 4 = 512 cols
  float4 q0 = *(const float4*)(Q + (size_t)g0 * NOUT + c);
  float4 q1 = *(const float4*)(Q + (size_t)g1 * NOUT + c);
  float4 cc = *(const float4*)(Cc + (size_t)b * NOUT + c);
  float4 o;
  o.x = fmaxf(0.5f * (q0.x + q1.x) + cc.x, 0.f);
  o.y = fmaxf(0.5f * (q0.y + q1.y) + cc.y, 0.f);
  o.z = fmaxf(0.5f * (q0.z + q1.z) + cc.z, 0.f);
  o.w = fmaxf(0.5f * (q0.w + q1.w) + cc.w, 0.f);
  *(float4*)(out + (size_t)row * NOUT + c) = o;
}

// ---------------------------------------------------------------------------
extern "C" void kernel_launch(void* const* d_in, const int* in_sizes, int n_in,
                              void* d_out, int out_size, void* d_ws,
                              size_t ws_size, hipStream_t stream) {
  (void)in_sizes; (void)n_in; (void)out_size; (void)ws_size;
  const float* OBJ = (const float*)d_in[0];  // [2048, 1024]
  const float* CTX = (const float*)d_in[1];  // [64, 1024]
  const int*   PR  = (const int*)d_in[2];    // [64, 256, 2]
  const float* W1  = (const float*)d_in[3];  // [2048, 1024]
  const float* B1  = (const float*)d_in[4];  // [1024]
  const float* W2  = (const float*)d_in[5];  // [1024, 512]
  const float* B2  = (const float*)d_in[6];  // [512]
  float* out = (float*)d_out;                // [16384, 512]

  // workspace layout (~15.2 MB)
  char* ws = (char*)d_ws;
  bf16*  W1b  = (bf16*)(ws + 0);                   // [2048,1024]  4 MB
  bf16*  OBJb = (bf16*)(ws + 4u  * 1024 * 1024);   // [2048,1024]  4 MB
  bf16*  W2t  = (bf16*)(ws + 8u  * 1024 * 1024);   // [512,1024]   1 MB (W2^T)
  bf16*  M1t  = (bf16*)(ws + 9u  * 1024 * 1024);   // [512,2048]   2 MB (M1^T)
  float* Q    = (float*)(ws + 11u * 1024 * 1024);  // [2048,512]   4 MB
  float* Cc   = (float*)(ws + 15u * 1024 * 1024);  // [64,512]     128 KB

  // 1) bf16 packs (B-operands transposed so GEMM staging is K-contiguous)
  cvt_f32_bf16<<<(2048 * 1024) / 1024, 256, 0, stream>>>(W1, W1b, 2048 * 1024);
  cvt_f32_bf16<<<(2048 * 1024) / 1024, 256, 0, stream>>>(OBJ, OBJb, 2048 * 1024);
  cvt_transpose_f32_bf16<<<(1024 * 512) / 256, 256, 0, stream>>>(W2, W2t, DDIM, NOUT);

  dim3 gA(NOUT / BN, 2048 / BM);  // (8, 16)
  // 2) M1^T = (W1 @ W2)^T   [512,2048] bf16
  gemm_bf16_wmma<true><<<gA, 256, 0, stream>>>(W1b, DDIM, W2t, DDIM,
                                               (void*)M1t, 2048, 2048, NOUT, DDIM);
  // 3) Q = OBJ @ M1_top   (B operand = M1t[:, 0:1024], row stride 2048)
  gemm_bf16_wmma<false><<<gA, 256, 0, stream>>>(OBJb, DDIM, M1t, 2048,
                                                (void*)Q, NOUT, 2048, NOUT, DDIM);
  // 4) Cc = CTX @ M1_bot + b1@W2 + b2
  ctx_bias_kernel<<<(BDIM * NOUT) / 256, 256, 0, stream>>>(CTX, B1, W2, B2,
                                                           M1t, Cc);
  // 5) gather + ReLU epilogue
  gather_relu_out<<<BDIM * RREL, 128, 0, stream>>>(PR, Q, Cc, out);
}
